// GPSODModel_82995948028331
// MI455X (gfx1250) — matrix operand order repaired and checked by reading, not compiled, not run on hardware
//
#include <hip/hip_runtime.h>
#include <hip/hip_bf16.h>
#include <math.h>

// ---------------- model dims ----------------
#define NNODE 4096
#define NEDGE 131072
#define IDIM  64
#define ED    32
#define HD    128
#define NHEAD 4
#define DHEAD 32
#define NLAYER 3

typedef __attribute__((ext_vector_type(16))) __bf16 v16bf;
typedef __attribute__((ext_vector_type(8)))  float  v8f;

union BF16x16 { v16bf v; unsigned u[8]; unsigned short s[16]; uint4 q[2]; };

__device__ __forceinline__ unsigned short f2bfu(float a) {
  unsigned ua = __builtin_bit_cast(unsigned, a);
  ua = (ua + 0x7FFFu + ((ua >> 16) & 1u)) >> 16;   // round-to-nearest-even
  return (unsigned short)ua;
}
__device__ __forceinline__ unsigned packbf(float a, float b) {
  return (unsigned)f2bfu(a) | ((unsigned)f2bfu(b) << 16);
}
__device__ __forceinline__ float bf2f(unsigned short s) {
  unsigned u = ((unsigned)s) << 16;
  return __builtin_bit_cast(float, u);
}

// xor-butterfly within each 16-lane row via v_permlane16_b32 (pure VALU,
// replaces ds_bpermute shuffles in the softmax hot loop)
__device__ __forceinline__ float pl16(float v, unsigned s0, unsigned s1) {
  unsigned u = __builtin_bit_cast(unsigned, v);
  u = __builtin_amdgcn_permlane16(u, u, s0, s1, false, false);
  return __builtin_bit_cast(float, u);
}
__device__ __forceinline__ float rowmax16(float v) {
  v = fmaxf(v, pl16(v, 0x67452301u, 0xEFCDAB89u));   // xor 1
  v = fmaxf(v, pl16(v, 0x54761032u, 0xDCFE98BAu));   // xor 2
  v = fmaxf(v, pl16(v, 0x32107654u, 0xBA98FEDCu));   // xor 4
  v = fmaxf(v, pl16(v, 0xFEDCBA98u, 0x76543210u));   // xor 8
  return v;
}
__device__ __forceinline__ float rowsum16(float v) {
  v += pl16(v, 0x67452301u, 0xEFCDAB89u);
  v += pl16(v, 0x54761032u, 0xDCFE98BAu);
  v += pl16(v, 0x32107654u, 0xBA98FEDCu);
  v += pl16(v, 0xFEDCBA98u, 0x76543210u);
  return v;
}

// =====================================================================
// Generic WMMA GEMM:  C[M,N] = act(A[M,K] @ W[N,K]^T + bias) (+ resid)
// W is PRE-CONVERTED bf16 -> inner loop is pure uint4 loads + WMMA.
// block = 128 threads = 4 waves; block tile 64x64; wave tile 16x64.
// A fragment (16x32 MxK): lane 0-15 -> M=lane, K in {kh..kh+7,16+kh..+7};
// B fragment (32x16 KxN): lane -> N=lane&15, K in {kh..kh+7,16+kh..+7}.
// ACT: 0 none, 1 relu.  ABF: A stored bf16.  Cf/Cb: fp32 / bf16 outputs
// (either may be null).  M,N multiples of 64; K multiple of 32.
// =====================================================================
template <int ACT, int ABF>
__global__ __launch_bounds__(128)
void gemm_wmma(const void* __restrict__ Ap, const unsigned short* __restrict__ Wb,
               const float* __restrict__ bias, const float* __restrict__ resid,
               float* __restrict__ Cf, unsigned short* __restrict__ Cb,
               int M, int N, int K)
{
  const int lane = threadIdx.x & 31;
  const int wave = threadIdx.x >> 5;
  const int row0 = blockIdx.x * 64 + wave * 16;
  const int col0 = blockIdx.y * 64;
  const int lr   = lane & 15;
  const int hi   = lane >> 4;
  const int kh   = hi * 8;
  const int klo  = hi * 16;
  const int arow = row0 + lr;

  v8f acc0 = {}, acc1 = {}, acc2 = {}, acc3 = {};

  for (int k0 = 0; k0 < K; k0 += 32) {
    BF16x16 af;
    if (ABF) {
      const unsigned short* A = (const unsigned short*)Ap;
      af.q[0] = *(const uint4*)(A + (size_t)arow * K + k0 + kh);
      af.q[1] = *(const uint4*)(A + (size_t)arow * K + k0 + 16 + kh);
    } else {
      const float* A  = (const float*)Ap;
      const float* q0 = A + (size_t)arow * K + k0 + kh;
      const float* q1 = A + (size_t)arow * K + k0 + 16 + kh;
#pragma unroll
      for (int j = 0; j < 4; ++j) {
        af.u[j]     = packbf(q0[2 * j], q0[2 * j + 1]);
        af.u[4 + j] = packbf(q1[2 * j], q1[2 * j + 1]);
      }
    }
#pragma unroll
    for (int t = 0; t < 4; ++t) {
      const unsigned short* wp = Wb + (size_t)(col0 + t * 16 + lr) * K + k0 + kh;
      BF16x16 bf;
      bf.q[0] = *(const uint4*)(wp);
      bf.q[1] = *(const uint4*)(wp + 16);
      v8f& acc = (t == 0) ? acc0 : (t == 1) ? acc1 : (t == 2) ? acc2 : acc3;
      acc = __builtin_amdgcn_wmma_f32_16x16x32_bf16(
          false, af.v, false, bf.v, (short)0, acc, false, false);
    }
  }

#pragma unroll
  for (int t = 0; t < 4; ++t) {
    v8f acc = (t == 0) ? acc0 : (t == 1) ? acc1 : (t == 2) ? acc2 : acc3;
    const int n = col0 + t * 16 + lr;
    const float bv = bias ? bias[n] : 0.f;
#pragma unroll
    for (int r = 0; r < 8; ++r) {
      const int m = row0 + hi * 8 + r;           // C layout: vgpr r -> M=r(+8)
      float v = acc[r] + bv;
      if (resid) v += resid[(size_t)m * N + n];
      if (ACT == 1) v = v > 0.f ? v : 0.f;
      if (Cf) Cf[(size_t)m * N + n] = v;
      if (Cb) Cb[(size_t)m * N + n] = f2bfu(v);
    }
  }
}

// =====================================================================
// Flash attention, one (64-query tile, head) per block, 4 waves.
// qkvb: [N, 3*HD] bf16 (q|k|v).  Ob: [N, HD] bf16 (pre-Wo).
// Per-chunk K/V staged cooperatively in LDS (V transposed) so fragment
// reads are aligned b128 LDS loads; softmax row-reductions use permlane16.
// =====================================================================
__global__ __launch_bounds__(128)
void attn_kernel(const unsigned short* __restrict__ qkvb,
                 unsigned short* __restrict__ Ob)
{
  __shared__ unsigned short klds[32 * 32];      // [key][dh] for this head
  __shared__ unsigned short vlds[32 * 32];      // [dh][key] (transposed)
  __shared__ unsigned short plds[4][16 * 32];   // per-wave P bounce buffer
  const int lane = threadIdx.x & 31;
  const int wave = threadIdx.x >> 5;
  const int head = blockIdx.y;
  const int q0   = blockIdx.x * 64 + wave * 16;
  const int lr   = lane & 15;
  const int hi   = lane >> 4;
  const int kh   = hi * 8;
  const int klo  = hi * 16;
  const int tkey = threadIdx.x >> 2;            // 0..31 (staging)
  const int tdh  = (threadIdx.x & 3) * 8;       // 0,8,16,24 (staging)
  const float scale = 0.17677669529663687f;     // 1/sqrt(32)

  BF16x16 qf;   // Q fragment (16 x 32), K-dim = DHEAD
  {
    const unsigned short* qp = qkvb + (size_t)(q0 + lr) * (3 * HD) + head * DHEAD;
    qf.q[0] = *(const uint4*)(qp + kh);
    qf.q[1] = *(const uint4*)(qp + 16 + kh);
  }

  v8f o0 = {}, o1 = {};
  float mrow[8], lrow[8];
#pragma unroll
  for (int r = 0; r < 8; ++r) { mrow[r] = -3.0e38f; lrow[r] = 0.f; }

  for (int kc = 0; kc < NNODE; kc += 32) {
    __syncthreads();                            // previous chunk fully consumed
    {
      const size_t rowb = (size_t)(kc + tkey) * (3 * HD) + head * DHEAD + tdh;
      const uint4 kq = *(const uint4*)(qkvb + rowb + HD);
      *(uint4*)&klds[tkey * 32 + tdh] = kq;
      const uint4 vq = *(const uint4*)(qkvb + rowb + 2 * HD);
      const unsigned short* vs = (const unsigned short*)&vq;
#pragma unroll
      for (int j = 0; j < 8; ++j) vlds[(tdh + j) * 32 + tkey] = vs[j];
    }
    __syncthreads();

    v8f lg0, lg1;
#pragma unroll
    for (int s = 0; s < 2; ++s) {
      BF16x16 kf;                               // B-frag: lane=key, e=dh
      kf.q[0] = *(const uint4*)&klds[(s * 16 + lr) * 32 + kh];
      kf.q[1] = *(const uint4*)&klds[(s * 16 + lr) * 32 + 16 + kh];
      v8f z = {};
      v8f res = __builtin_amdgcn_wmma_f32_16x16x32_bf16(
          false, qf.v, false, kf.v, (short)0, z, false, false);
      if (s == 0) lg0 = res; else lg1 = res;
    }
    // online softmax over this 32-key chunk (VALU permlane reductions)
#pragma unroll
    for (int r = 0; r < 8; ++r) {
      float v0 = lg0[r] * scale, v1 = lg1[r] * scale;
      const float mx = rowmax16(fmaxf(v0, v1));
      const float newm = fmaxf(mrow[r], mx);
      const float sf = __expf(mrow[r] - newm);
      o0[r] *= sf; o1[r] *= sf; lrow[r] *= sf;
      mrow[r] = newm;
      const float p0 = __expf(v0 - newm);
      const float p1 = __expf(v1 - newm);
      lrow[r] += rowsum16(p0 + p1);
      const int m = hi * 8 + r;                 // logits C-layout row
      plds[wave][m * 32 + lr]      = f2bfu(p0);
      plds[wave][m * 32 + 16 + lr] = f2bfu(p1);
    }
    asm volatile("s_wait_dscnt 0" ::: "memory");  // P stores visible to wave
    BF16x16 pf;                                 // A-frag: lane=M, K striped
    {
      const unsigned short* pp = &plds[wave][lr * 32];
      pf.q[0] = *(const uint4*)(pp + kh);
      pf.q[1] = *(const uint4*)(pp + 16 + kh);
    }
    // O += P @ V   (two 16x16 output tiles covering DHEAD=32)
#pragma unroll
    for (int t = 0; t < 2; ++t) {
      BF16x16 vf;                               // B-frag from transposed V
      vf.q[0] = *(const uint4*)&vlds[(t * 16 + lr) * 32 + kh];
      vf.q[1] = *(const uint4*)&vlds[(t * 16 + lr) * 32 + 16 + kh];
      v8f& oa = t ? o1 : o0;
      oa = __builtin_amdgcn_wmma_f32_16x16x32_bf16(
          false, pf.v, false, vf.v, (short)0, oa, false, false);
    }
  }
#pragma unroll
  for (int t = 0; t < 2; ++t) {
    v8f oa = t ? o1 : o0;
#pragma unroll
    for (int r = 0; r < 8; ++r) {
      const int m = q0 + hi * 8 + r;
      const int c = head * DHEAD + t * 16 + lr;
      Ob[(size_t)m * HD + c] = f2bfu(oa[r] / lrow[r]);
    }
  }
}

// ---------------- GINE message + scatter-add ----------------
__global__ __launch_bounds__(256)
void edge_msg_kernel(const float* __restrict__ h,
                     const unsigned short* __restrict__ ee,
                     const int* __restrict__ src, const int* __restrict__ dst,
                     float* __restrict__ aggr)
{
  const int idx = blockIdx.x * 256 + threadIdx.x;   // NEDGE*32 threads
  const int e = idx >> 5;
  const int c = (idx & 31) << 2;
  const int s = src[e], d = dst[e];
  const float* hp = h + (size_t)s * HD + c;
  const unsigned short* ep = ee + (size_t)e * HD + c;
  float* ap = aggr + (size_t)d * HD + c;
#pragma unroll
  for (int j = 0; j < 4; ++j) {
    float v = hp[j] + bf2f(ep[j]);
    v = v > 0.f ? v : 0.f;
    atomicAdd(ap + j, v);
  }
}

__global__ void zero_kernel(float* __restrict__ p, int n) {
  const int i = blockIdx.x * 256 + threadIdx.x;
  if (i < n) p[i] = 0.f;
}

// c_f32 / c_bf16 outputs; either may be null
__global__ void add2_kernel(const float* __restrict__ a,
                            const float* __restrict__ b,
                            float* __restrict__ cf, unsigned short* __restrict__ cb,
                            int n) {
  const int i = blockIdx.x * 256 + threadIdx.x;
  if (i < n) {
    const float v = a[i] + b[i];
    if (cf) cf[i] = v;
    if (cb) cb[i] = f2bfu(v);
  }
}

__global__ void cvt_kernel(const float* __restrict__ a,
                           unsigned short* __restrict__ b, int n) {
  const int i = blockIdx.x * 256 + threadIdx.x;
  if (i < n) b[i] = f2bfu(a[i]);
}

// ---------------- GraphNorm: one column per block, dual output ----------------
__global__ __launch_bounds__(256)
void gnorm_kernel(const float* __restrict__ X, float* __restrict__ Hout,
                  unsigned short* __restrict__ Houtb,
                  const float* __restrict__ gamma, const float* __restrict__ beta,
                  const float* __restrict__ alpha)
{
  __shared__ float red[256];
  const int c = blockIdx.x;
  const int tid = threadIdx.x;
  float s = 0.f;
  for (int r = tid; r < NNODE; r += 256) s += X[(size_t)r * HD + c];
  red[tid] = s; __syncthreads();
  for (int st = 128; st > 0; st >>= 1) {
    if (tid < st) red[tid] += red[tid + st];
    __syncthreads();
  }
  const float mn = red[0] / (float)NNODE;
  __syncthreads();
  const float sub = alpha[0] * mn;
  float sq = 0.f;
  for (int r = tid; r < NNODE; r += 256) {
    const float d = X[(size_t)r * HD + c] - sub;
    sq += d * d;
  }
  red[tid] = sq; __syncthreads();
  for (int st = 128; st > 0; st >>= 1) {
    if (tid < st) red[tid] += red[tid + st];
    __syncthreads();
  }
  const float var = red[0] / (float)NNODE;
  const float inv = gamma[c] / (sqrtf(var) + 1e-5f);
  const float bt = beta[c];
  for (int r = tid; r < NNODE; r += 256) {
    const float v = (X[(size_t)r * HD + c] - sub) * inv + bt;
    Hout[(size_t)r * HD + c] = v;
    Houtb[(size_t)r * HD + c] = f2bfu(v);
  }
}

// =====================================================================
extern "C" void kernel_launch(void* const* d_in, const int* in_sizes, int n_in,
                              void* d_out, int out_size, void* d_ws, size_t ws_size,
                              hipStream_t stream)
{
  (void)in_sizes; (void)n_in; (void)out_size; (void)ws_size;
  // inputs flattened in setup_inputs() dict insertion order
  const float* x        = (const float*)d_in[0];
  const float* ea       = (const float*)d_in[1];
  const float* np_W1    = (const float*)d_in[2];
  const float* np_b1    = (const float*)d_in[3];
  const float* np_W2    = (const float*)d_in[4];
  const float* np_b2    = (const float*)d_in[5];
  const float* ep_W1    = (const float*)d_in[6];
  const float* ep_b1    = (const float*)d_in[7];
  const float* ep_W2    = (const float*)d_in[8];
  const float* ep_b2    = (const float*)d_in[9];
  const float* gine_W1  = (const float*)d_in[10];
  const float* gine_b1  = (const float*)d_in[11];
  const float* gine_W2  = (const float*)d_in[12];
  const float* gine_b2  = (const float*)d_in[13];
  const float* attn_Wqkv= (const float*)d_in[14];
  const float* attn_bqkv= (const float*)d_in[15];
  const float* attn_Wo  = (const float*)d_in[16];
  const float* attn_bo  = (const float*)d_in[17];
  const float* mlp_W1   = (const float*)d_in[18];
  const float* mlp_b1   = (const float*)d_in[19];
  const float* mlp_W2   = (const float*)d_in[20];
  const float* mlp_b2   = (const float*)d_in[21];
  const float* gn_gamma = (const float*)d_in[22];
  const float* gn_beta  = (const float*)d_in[23];
  const float* gn_alpha = (const float*)d_in[24];
  const float* dec_W    = (const float*)d_in[25];
  const int*   eidx     = (const int*)d_in[26];
  const int* src = eidx;
  const int* dst = eidx + NEDGE;

  // workspace: [ee bf16 32MB][wbuf 2MB][pool 32MB] = 66MB
  const size_t MB = 1u << 20;
  unsigned short* ee = (unsigned short*)d_ws;
  unsigned short* wbuf = (unsigned short*)((char*)d_ws + (size_t)NEDGE * HD * 2);
  char* pool = (char*)d_ws + (size_t)NEDGE * HD * 2 + 2 * MB;

  // converted weights (element offsets in wbuf)
  unsigned short* wb_np1 = wbuf;                         // 128*64
  unsigned short* wb_np2 = wb_np1 + 8192;                // 128*128
  unsigned short* wb_ep1 = wb_np2 + 16384;               // 128*32
  unsigned short* wb_ep2 = wb_ep1 + 4096;                // 128*128
  unsigned short* wb_gi1 = wb_ep2 + 16384;               // 3*128*128
  unsigned short* wb_gi2 = wb_gi1 + 49152;
  unsigned short* wb_qkv = wb_gi2 + 49152;               // 3*384*128
  unsigned short* wb_wo  = wb_qkv + 147456;              // 3*128*128
  unsigned short* wb_m1  = wb_wo  + 49152;               // 3*256*128
  unsigned short* wb_m2  = wb_m1  + 98304;               // 3*128*256
  unsigned short* wb_dec = wb_m2  + 98304;               // 128*128

  // pool: e1bf (32MB, edge phase) overlays node-phase buffers
  unsigned short* e1bf   = (unsigned short*)pool;
  float*          h      = (float*)(pool + 0 * MB);
  unsigned short* hbf    = (unsigned short*)(pool + 2 * MB);
  unsigned short* tmpbf  = (unsigned short*)(pool + 3 * MB);
  float*          aggr   = (float*)(pool + 4 * MB);
  unsigned short* hinbf  = (unsigned short*)(pool + 6 * MB);
  unsigned short* qkvbf  = (unsigned short*)(pool + 7 * MB);   // 3MB
  unsigned short* obf    = (unsigned short*)(pool + 10 * MB);
  float*          hl     = (float*)(pool + 11 * MB);
  float*          ha     = (float*)(pool + 13 * MB);
  float*          gout   = (float*)(pool + 15 * MB);
  unsigned short* goutbf = (unsigned short*)(pool + 17 * MB);
  unsigned short* t1bf   = (unsigned short*)(pool + 18 * MB);  // 2MB
  float*          xb     = (float*)(pool + 20 * MB);
  unsigned short* tbf    = (unsigned short*)(pool + 22 * MB);

  const dim3 blk(128);
  const int NB = NNODE / 64;                         // 64
  auto cvt = [&](const float* a, unsigned short* b, int n) {
    cvt_kernel<<<(n + 255) / 256, 256, 0, stream>>>(a, b, n);
  };

  // ---- pre-convert all weights to bf16 (once per launch) ----
  cvt(np_W1, wb_np1, 8192);    cvt(np_W2, wb_np2, 16384);
  cvt(ep_W1, wb_ep1, 4096);    cvt(ep_W2, wb_ep2, 16384);
  cvt(gine_W1, wb_gi1, 49152); cvt(gine_W2, wb_gi2, 49152);
  cvt(attn_Wqkv, wb_qkv, 147456); cvt(attn_Wo, wb_wo, 49152);
  cvt(mlp_W1, wb_m1, 98304);   cvt(mlp_W2, wb_m2, 98304);
  cvt(dec_W, wb_dec, 16384);

  // ---- edge_proj (e1bf dead before node phase) ----
  gemm_wmma<1,0><<<dim3(NEDGE/64, HD/64), blk, 0, stream>>>(ea,   wb_ep1, ep_b1, nullptr, nullptr, e1bf, NEDGE, HD, ED);
  gemm_wmma<0,1><<<dim3(NEDGE/64, HD/64), blk, 0, stream>>>(e1bf, wb_ep2, ep_b2, nullptr, nullptr, ee,   NEDGE, HD, HD);
  // ---- node_proj ----
  gemm_wmma<1,0><<<dim3(NB, HD/64), blk, 0, stream>>>(x,     wb_np1, np_b1, nullptr, nullptr, tmpbf, NNODE, HD, IDIM);
  gemm_wmma<0,1><<<dim3(NB, HD/64), blk, 0, stream>>>(tmpbf, wb_np2, np_b2, nullptr, h, hbf, NNODE, HD, HD);

  for (int i = 0; i < NLAYER; ++i) {
    // GINEConv
    zero_kernel<<<NNODE * HD / 256, 256, 0, stream>>>(aggr, NNODE * HD);
    edge_msg_kernel<<<NEDGE * 32 / 256, 256, 0, stream>>>(h, ee, src, dst, aggr);
    add2_kernel<<<NNODE * HD / 256, 256, 0, stream>>>(h, aggr, nullptr, hinbf, NNODE * HD);
    gemm_wmma<1,1><<<dim3(NB, 2), blk, 0, stream>>>(hinbf, wb_gi1 + (size_t)i*HD*HD, gine_b1 + i*HD, nullptr, nullptr, tmpbf, NNODE, HD, HD);
    gemm_wmma<0,1><<<dim3(NB, 2), blk, 0, stream>>>(tmpbf, wb_gi2 + (size_t)i*HD*HD, gine_b2 + i*HD, h, hl, nullptr, NNODE, HD, HD);
    // attention
    gemm_wmma<0,1><<<dim3(NB, 6), blk, 0, stream>>>(hbf, wb_qkv + (size_t)i*3*HD*HD, attn_bqkv + i*3*HD, nullptr, nullptr, qkvbf, NNODE, 3*HD, HD);
    attn_kernel<<<dim3(NB, NHEAD), blk, 0, stream>>>(qkvbf, obf);
    gemm_wmma<0,1><<<dim3(NB, 2), blk, 0, stream>>>(obf, wb_wo + (size_t)i*HD*HD, attn_bo + i*HD, h, ha, nullptr, NNODE, HD, HD);
    // combine + FFN
    add2_kernel<<<NNODE * HD / 256, 256, 0, stream>>>(hl, ha, gout, goutbf, NNODE * HD);
    gemm_wmma<1,1><<<dim3(NB, 4), blk, 0, stream>>>(goutbf, wb_m1 + (size_t)i*2*HD*HD, mlp_b1 + i*2*HD, nullptr, nullptr, t1bf, NNODE, 2*HD, HD);
    gemm_wmma<0,1><<<dim3(NB, 2), blk, 0, stream>>>(t1bf, wb_m2 + (size_t)i*2*HD*HD, mlp_b2 + i*HD, gout, xb, nullptr, NNODE, HD, 2*HD);
    // GraphNorm
    gnorm_kernel<<<HD, 256, 0, stream>>>(xb, h, hbf, gn_gamma + i*HD, gn_beta + i*HD, gn_alpha + i);
  }

  // ---- bilinear decoder: t = h @ dec_W^T;  od = h @ t^T ----
  gemm_wmma<0,1><<<dim3(NB, 2), blk, 0, stream>>>(hbf, wb_dec, nullptr, nullptr, nullptr, tbf, NNODE, HD, HD);
  gemm_wmma<0,1><<<dim3(NB, NB), blk, 0, stream>>>(hbf, tbf, nullptr, nullptr, (float*)d_out, nullptr, NNODE, NNODE, HD);
}
